// PointTransformerSeg_86930138071566
// MI455X (gfx1250) — matrix operand order repaired and checked by reading, not compiled, not run on hardware
//
#include <hip/hip_runtime.h>

// ---------------------------------------------------------------------------
// Types for WMMA (gfx1250, wave32)
// ---------------------------------------------------------------------------
typedef __attribute__((ext_vector_type(16))) __bf16 bf16x16;
typedef __attribute__((ext_vector_type(8)))  float  f32x8;

#define TBM 32
#define TBN 64
#define TBK 64   // two 32-K slices per iteration -> 2 WMMAs per wave per barrier

static __device__ __forceinline__ float bnc() { return 0.99999500003749973f; } // 1/sqrt(1+1e-5)

static __device__ __forceinline__ unsigned long long pack4bf(float a, float b, float c, float d)
{
    union { __bf16 h[4]; unsigned long long u; } p;
    p.h[0] = (__bf16)a; p.h[1] = (__bf16)b; p.h[2] = (__bf16)c; p.h[3] = (__bf16)d;
    return p.u;
}

// ---------------------------------------------------------------------------
// GEMM: C[M,N] = epilogue(A[M,K] @ W[K,N])
// epilogue: +bias -> BN(eval: *gamma*bnc + beta) -> ReLU -> +res
// A, W fp32 in memory; staged to LDS as bf16 IN FRAGMENT ORDER so each lane's
// WMMA operand is one contiguous 32B vector load (ds_load_b128 x2).
// Block: 256 threads = 8 waves covering a 32x64 tile (2x4 grid of 16x16 WMMA tiles)
// Block-uniform fast path for fully-interior tiles (no per-element guards);
// guarded slow path only for edge tiles.
//
// Fragment-order LDS layout (per 32-K slice):
//   A: region [slice*1024 + wm*512 + lane*16 + e]  (bf16 elems)
//        lane = (m&15) + 16*((k&31)>>3 & 1),  e = (k&7) + 8*((k&31)>>4)
//   B: region [slice*2048 + wn*512 + lane*16 + e]
//        lane = (n&15) + 16*((k&31)>>3 & 1),  e = (k&7) + 8*((k&31)>>4)
// ---------------------------------------------------------------------------
__global__ __launch_bounds__(256) void gemm_wmma(
    const float* __restrict__ A, const float* __restrict__ W,
    const float* __restrict__ bias,
    const float* __restrict__ bng, const float* __restrict__ bnb,
    const float* __restrict__ res,
    float* __restrict__ C, int M, int N, int K, int relu)
{
    __shared__ __align__(16) __bf16 sA[2 * 1024];  // 32 x 64 bf16, fragment order
    __shared__ __align__(16) __bf16 sB[2 * 2048];  // 64 x 64 bf16, fragment order

    const int tid  = threadIdx.x;
    const int lane = tid & 31;
    const int wave = tid >> 5;       // 0..7
    const int wm   = wave >> 2;      // 0..1
    const int wn   = wave & 3;       // 0..3
    const int bm0  = blockIdx.y * TBM;
    const int bn0  = blockIdx.x * TBN;
    const int khalf = lane >> 4;     // 0/1
    const int l15   = lane & 15;

    const bool interior = (bm0 + TBM <= M) && (bn0 + TBN <= N);

    f32x8 acc = {};

    for (int k0 = 0; k0 < K; k0 += TBK) {
        const bool full = interior && (k0 + TBK <= K);   // uniform across block

        if (full) {
            // ---------------- fast path: no guards ----------------
            for (int it = tid; it < 512; it += 256) {
                int r = it >> 4;            // 0..31 tile row
                int kbase = (it & 15) << 2; // 0,4,...,60
                long ro = (long)(bm0 + r) * K + (k0 + kbase);
                float v0 = A[ro + 0], v1 = A[ro + 1], v2 = A[ro + 2], v3 = A[ro + 3];
                int s  = kbase >> 5;
                int kk = kbase & 31;
                int ln = (r & 15) + (((kk >> 3) & 1) << 4);
                int e0 = (kk & 7) + ((kk >> 4) << 3);
                int off = s * 1024 + ((r >> 4) << 9) + (ln << 4) + e0;
                *(unsigned long long*)(&sA[off]) = pack4bf(v0, v1, v2, v3);
            }
            for (int it = tid; it < 1024; it += 256) {
                int n = it >> 4;            // 0..63 tile col
                int kbase = (it & 15) << 2; // 0..60
                long co = (long)(k0 + kbase) * N + (bn0 + n);
                float v0 = W[co + 0L * N], v1 = W[co + 1L * N];
                float v2 = W[co + 2L * N], v3 = W[co + 3L * N];
                int s  = kbase >> 5;
                int kk = kbase & 31;
                int ln = (n & 15) + (((kk >> 3) & 1) << 4);
                int e0 = (kk & 7) + ((kk >> 4) << 3);
                int off = s * 2048 + ((n >> 4) << 9) + (ln << 4) + e0;
                *(unsigned long long*)(&sB[off]) = pack4bf(v0, v1, v2, v3);
            }
            // prefetch next K tile (only when it is also fully in-bounds)
            if (k0 + 2 * TBK <= K) {
                __builtin_prefetch(&A[(long)(bm0 + lane) * K + (k0 + TBK) + (wave << 3)], 0, 1);
                __builtin_prefetch(&W[(long)(k0 + TBK + (tid & 63)) * N + bn0 + ((tid >> 6) << 4)], 0, 1);
            }
        } else {
            // ---------------- guarded path: edge tiles ----------------
            for (int it = tid; it < 512; it += 256) {
                int r = it >> 4;
                int kbase = (it & 15) << 2;
                int gr = bm0 + r;
                float v0 = 0.f, v1 = 0.f, v2 = 0.f, v3 = 0.f;
                if (gr < M) {
                    int kr = k0 + kbase;
                    long ro = (long)gr * K + kr;
                    int rem = K - kr;
                    if (rem >= 4) { v0 = A[ro]; v1 = A[ro + 1]; v2 = A[ro + 2]; v3 = A[ro + 3]; }
                    else {
                        if (rem > 0) v0 = A[ro];
                        if (rem > 1) v1 = A[ro + 1];
                        if (rem > 2) v2 = A[ro + 2];
                    }
                }
                int s  = kbase >> 5;
                int kk = kbase & 31;
                int ln = (r & 15) + (((kk >> 3) & 1) << 4);
                int e0 = (kk & 7) + ((kk >> 4) << 3);
                int off = s * 1024 + ((r >> 4) << 9) + (ln << 4) + e0;
                *(unsigned long long*)(&sA[off]) = pack4bf(v0, v1, v2, v3);
            }
            for (int it = tid; it < 1024; it += 256) {
                int n = it >> 4;
                int kbase = (it & 15) << 2;
                int gc = bn0 + n;
                float v0 = 0.f, v1 = 0.f, v2 = 0.f, v3 = 0.f;
                if (gc < N) {
                    int kr = k0 + kbase;
                    if (kr + 0 < K) v0 = W[(long)(kr + 0) * N + gc];
                    if (kr + 1 < K) v1 = W[(long)(kr + 1) * N + gc];
                    if (kr + 2 < K) v2 = W[(long)(kr + 2) * N + gc];
                    if (kr + 3 < K) v3 = W[(long)(kr + 3) * N + gc];
                }
                int s  = kbase >> 5;
                int kk = kbase & 31;
                int ln = (n & 15) + (((kk >> 3) & 1) << 4);
                int e0 = (kk & 7) + ((kk >> 4) << 3);
                int off = s * 2048 + ((n >> 4) << 9) + (ln << 4) + e0;
                *(unsigned long long*)(&sB[off]) = pack4bf(v0, v1, v2, v3);
            }
        }
        __syncthreads();

        // ---- two WMMAs (K slices 0 and 1); operands are contiguous 32B loads ----
        {
            const bf16x16 a0 = *(const bf16x16*)&sA[(wm << 9) + (lane << 4)];
            const bf16x16 b0 = *(const bf16x16*)&sB[(wn << 9) + (lane << 4)];
            acc = __builtin_amdgcn_wmma_f32_16x16x32_bf16(
                false, a0, false, b0, (short)0, acc, false, false);
            const bf16x16 a1 = *(const bf16x16*)&sA[1024 + (wm << 9) + (lane << 4)];
            const bf16x16 b1 = *(const bf16x16*)&sB[2048 + (wn << 9) + (lane << 4)];
            acc = __builtin_amdgcn_wmma_f32_16x16x32_bf16(
                false, a1, false, b1, (short)0, acc, false, false);
        }
        __syncthreads();
    }

    // Epilogue. C layout: VGPR r -> row (r + 8*khalf), col = l15.
    const int cn = bn0 + wn * 16 + l15;
#pragma unroll
    for (int r = 0; r < 8; ++r) {
        int cm = bm0 + wm * 16 + r + 8 * khalf;
        if (cm < M && cn < N) {
            float v = acc[r];
            if (bias) v += bias[cn];
            if (bng)  v = v * (bng[cn] * bnc()) + bnb[cn];
            if (relu) v = fmaxf(v, 0.0f);
            if (res)  v += res[(long)cm * N + cn];
            C[(long)cm * N + cn] = v;
        }
    }
}

// ---------------------------------------------------------------------------
// Farthest point sampling (single workgroup, deterministic start at 0)
// ---------------------------------------------------------------------------
__global__ __launch_bounds__(1024) void fps_kernel(const float* __restrict__ xyz,
                                                   int n, int npoint, int* __restrict__ out)
{
    __shared__ float sdist[4096];
    __shared__ float rv[1024];
    __shared__ int   ri[1024];
    __shared__ int   sfar;
    const int t = threadIdx.x;
    for (int i = t; i < n; i += 1024) sdist[i] = 1e10f;
    if (t == 0) sfar = 0;
    __syncthreads();
    for (int it = 0; it < npoint; ++it) {
        int far = sfar;
        if (t == 0) out[it] = far;
        float cx = xyz[far * 3 + 0], cy = xyz[far * 3 + 1], cz = xyz[far * 3 + 2];
        float bv = -1.0f; int bi = 0x7fffffff;
        for (int i = t; i < n; i += 1024) {
            float dx = xyz[i * 3 + 0] - cx;
            float dy = xyz[i * 3 + 1] - cy;
            float dz = xyz[i * 3 + 2] - cz;
            float d  = dx * dx + dy * dy + dz * dz;
            float nd = fminf(sdist[i], d);
            sdist[i] = nd;
            if (nd > bv) { bv = nd; bi = i; }
        }
        rv[t] = bv; ri[t] = bi;
        __syncthreads();
        for (int s = 512; s > 0; s >>= 1) {
            if (t < s) {
                if (rv[t + s] > rv[t] || (rv[t + s] == rv[t] && ri[t + s] < ri[t])) {
                    rv[t] = rv[t + s]; ri[t] = ri[t + s];
                }
            }
            __syncthreads();
        }
        if (t == 0) sfar = ri[0];
        __syncthreads();
    }
}

// ---------------------------------------------------------------------------
// Brute-force kNN (k <= 16), sorted ascending; optional squared distances out
// ---------------------------------------------------------------------------
__global__ void knn_kernel(const float* __restrict__ q, int nq,
                           const float* __restrict__ ref, int nref, int k,
                           int* __restrict__ idx, float* __restrict__ odist)
{
    int i = blockIdx.x * blockDim.x + threadIdx.x;
    if (i >= nq) return;
    float bd[16]; int bi[16];
    for (int j = 0; j < k; ++j) { bd[j] = 3.4e38f; bi[j] = 0; }
    float qx = q[i * 3 + 0], qy = q[i * 3 + 1], qz = q[i * 3 + 2];
    for (int j = 0; j < nref; ++j) {
        float dx = ref[j * 3 + 0] - qx;
        float dy = ref[j * 3 + 1] - qy;
        float dz = ref[j * 3 + 2] - qz;
        float d  = dx * dx + dy * dy + dz * dz;
        if (d < bd[k - 1]) {
            int p = k - 1;
            while (p > 0 && bd[p - 1] > d) { bd[p] = bd[p - 1]; bi[p] = bi[p - 1]; --p; }
            bd[p] = d; bi[p] = j;
        }
    }
    for (int j = 0; j < k; ++j) {
        idx[(long)i * k + j] = bi[j];
        if (odist) odist[(long)i * k + j] = bd[j];
    }
}

// gather rows: dst[r, :] = src[idx[r], :]
__global__ void gather_rows(const float* __restrict__ src, const int* __restrict__ idx,
                            int n, int c, float* __restrict__ dst)
{
    long e = (long)blockIdx.x * blockDim.x + threadIdx.x;
    if (e >= (long)n * c) return;
    int r = (int)(e / c), ch = (int)(e % c);
    dst[e] = src[(long)idx[r] * c + ch];
}

// h[s,k,:] = concat(xyz[idx]-newxyz[s], pts[idx])
__global__ void build_sa_h(const float* __restrict__ xyz, const float* __restrict__ nxyz,
                           const float* __restrict__ pts, const int* __restrict__ idx,
                           int S, int kn, int C, float* __restrict__ h)
{
    int cin = 3 + C;
    long tot = (long)S * kn * cin;
    long e = (long)blockIdx.x * blockDim.x + threadIdx.x;
    if (e >= tot) return;
    int row = (int)(e / cin), ch = (int)(e % cin);
    int s = row / kn;
    int j = idx[row];
    float v;
    if (ch < 3) v = xyz[(long)j * 3 + ch] - nxyz[(long)s * 3 + ch];
    else        v = pts[(long)j * C + (ch - 3)];
    h[e] = v;
}

// out[s,c] = max_k h[s,k,c]
__global__ void maxpool_k(const float* __restrict__ h, int S, int kn, int C,
                          float* __restrict__ out)
{
    long e = (long)blockIdx.x * blockDim.x + threadIdx.x;
    if (e >= (long)S * C) return;
    int s = (int)(e / C), ch = (int)(e % C);
    float m = -3.4e38f;
    for (int j = 0; j < kn; ++j) m = fmaxf(m, h[((long)s * kn + j) * C + ch]);
    out[e] = m;
}

// posin[row,:] = xyz[r0 + row/kn] - xyz[idx[row]]   (idx pre-offset by r0*kn)
__global__ void build_posin(const float* __restrict__ xyz, const int* __restrict__ idx,
                            int r0, int rows, int kn, float* __restrict__ posin)
{
    long tot = (long)rows * kn * 3;
    long e = (long)blockIdx.x * blockDim.x + threadIdx.x;
    if (e >= tot) return;
    int row = (int)(e / 3), ch = (int)(e % 3);
    int p = r0 + row / kn;
    int j = idx[row];
    posin[e] = xyz[(long)p * 3 + ch] - xyz[(long)j * 3 + ch];
}

// attn_in[row,f] = q[r0+row/kn, f] - kf[idx[row], f] + pos[row, f]
__global__ void tb_prep(const float* __restrict__ q, const float* __restrict__ kf,
                        const float* __restrict__ pos, const int* __restrict__ idx,
                        int r0, int rows, int kn, int d, float* __restrict__ attn_in)
{
    long tot = (long)rows * kn * d;
    long e = (long)blockIdx.x * blockDim.x + threadIdx.x;
    if (e >= tot) return;
    int row = (int)(e / d), f = (int)(e % d);
    int p = r0 + row / kn;
    int j = idx[row];
    attn_in[e] = q[(long)p * d + f] - kf[(long)j * d + f] + pos[e];
}

// agg[r0+p, f] = sum_k softmax_k(attn[p,k,f]*scale) * (vf[idx[p,k], f] + pos[p,k,f])
__global__ void attn_agg(const float* __restrict__ attn, const float* __restrict__ pos,
                         const float* __restrict__ vf, const int* __restrict__ idx,
                         int r0, int rows, int kn, int d, float scale,
                         float* __restrict__ agg)
{
    long tot = (long)rows * d;
    long e = (long)blockIdx.x * blockDim.x + threadIdx.x;
    if (e >= tot) return;
    int p = (int)(e / d), f = (int)(e % d);
    float mx = -3.4e38f;
    for (int k = 0; k < kn; ++k)
        mx = fmaxf(mx, attn[((long)p * kn + k) * d + f] * scale);
    float s = 0.0f, accv = 0.0f;
    for (int k = 0; k < kn; ++k) {
        float w = __expf(attn[((long)p * kn + k) * d + f] * scale - mx);
        int j = idx[(long)p * kn + k];
        s += w;
        accv += w * (vf[(long)j * d + f] + pos[((long)p * kn + k) * d + f]);
    }
    agg[(long)(r0 + p) * d + f] = accv / s;
}

// out[i,c] = sum_j (1/(d_j+1e-8))/W * src[idx[i,j], c]  (+ addv)
__global__ void interp3_k(const int* __restrict__ idx, const float* __restrict__ dist,
                          const float* __restrict__ src, const float* __restrict__ addv,
                          int nt, int c, float* __restrict__ out)
{
    long e = (long)blockIdx.x * blockDim.x + threadIdx.x;
    if (e >= (long)nt * c) return;
    int i = (int)(e / c), ch = (int)(e % c);
    float w0 = 1.0f / (dist[(long)i * 3 + 0] + 1e-8f);
    float w1 = 1.0f / (dist[(long)i * 3 + 1] + 1e-8f);
    float w2 = 1.0f / (dist[(long)i * 3 + 2] + 1e-8f);
    float s = w0 + w1 + w2;
    float v = (w0 * src[(long)idx[(long)i * 3 + 0] * c + ch] +
               w1 * src[(long)idx[(long)i * 3 + 1] * c + ch] +
               w2 * src[(long)idx[(long)i * 3 + 2] * c + ch]) / s;
    if (addv) v += addv[e];
    out[e] = v;
}

// ---------------------------------------------------------------------------
// Host orchestration
// ---------------------------------------------------------------------------
static inline int cdiv(long a, long b) { return (int)((a + b - 1) / b); }

struct Bump {
    char* base; size_t off, cap;
    void* raw(size_t bytes) {
        size_t o = (off + 255) & ~(size_t)255;
        off = o + bytes;
        return base + o;
    }
    float* f(size_t n) { return (float*)raw(n * sizeof(float)); }
    int*   i(size_t n) { return (int*)raw(n * sizeof(int)); }
};

struct Lin { const float* W; const float* b; };
struct TB  { Lin d1, d2, fc1, fc2, g1, g2; const float* wk; const float* wq; const float* wv; };
struct SAL { const float* beta; const float* gamma; Lin lin; };
struct TU  { const float* bn1b; const float* bn1g; const float* bn2b; const float* bn2g; Lin fc1, fc2; };

static inline const float* FP(void* const* din, int& p) { return (const float*)din[p++]; }
static inline Lin rdLin(void* const* din, int& p)   { Lin l; l.W = FP(din, p); l.b = FP(din, p); return l; }
static inline TB  rdTB(void* const* din, int& p) {
    TB t;
    t.d1 = rdLin(din, p); t.d2 = rdLin(din, p);
    t.fc1 = rdLin(din, p); t.fc2 = rdLin(din, p);
    t.g1 = rdLin(din, p); t.g2 = rdLin(din, p);
    t.wk = FP(din, p); t.wq = FP(din, p); t.wv = FP(din, p); // sorted: wk < wq < wv
    return t;
}
static inline SAL rdSAL(void* const* din, int& p) {
    SAL s; s.beta = FP(din, p); s.gamma = FP(din, p); s.lin = rdLin(din, p); return s;
}
static inline TU rdTU(void* const* din, int& p) {
    TU t;
    t.bn1b = FP(din, p); t.bn1g = FP(din, p);
    t.bn2b = FP(din, p); t.bn2g = FP(din, p);
    t.fc1 = rdLin(din, p); t.fc2 = rdLin(din, p);
    return t;
}

static void gemm(hipStream_t st, const float* A, const float* W, const float* bias,
                 float* C, int M, int N, int K, int relu,
                 const float* bng = nullptr, const float* bnb = nullptr,
                 const float* res = nullptr)
{
    dim3 g(cdiv(N, TBN), cdiv(M, TBM));
    gemm_wmma<<<g, 256, 0, st>>>(A, W, bias, bng, bnb, res, C, M, N, K, relu);
}
static void gemmL(hipStream_t st, const float* A, const Lin& L, float* C,
                  int M, int N, int K, int relu,
                  const float* bng = nullptr, const float* bnb = nullptr,
                  const float* res = nullptr)
{
    gemm(st, A, L.W, L.b, C, M, N, K, relu, bng, bnb, res);
}

#define EW(kname, tot, ...) kname<<<cdiv((tot), 256), 256, 0, st>>>(__VA_ARGS__)

// Point Transformer vector self-attention block, in-place on pts (n, C)
static void run_tblock(hipStream_t st, Bump s, const TB& t,
                       const float* xyz, float* pts, int n, int C, int kn)
{
    const int D = 512;
    const float scale = 0.04419417382415922f;  // 1/sqrt(512)

    int* idx = s.i((size_t)n * kn);
    knn_kernel<<<cdiv(n, 128), 128, 0, st>>>(xyz, n, xyz, n, kn, idx, nullptr);

    float* xb = s.f((size_t)n * D);
    float* qb = s.f((size_t)n * D);
    float* kb = s.f((size_t)n * D);
    float* vb = s.f((size_t)n * D);
    float* agg = s.f((size_t)n * D);

    gemmL(st, pts, t.fc1, xb, n, D, C, 0);
    gemm(st, xb, t.wq, nullptr, qb, n, D, D, 0);
    gemm(st, xb, t.wk, nullptr, kb, n, D, D, 0);
    gemm(st, xb, t.wv, nullptr, vb, n, D, D, 0);

    const int CH = 1024;
    float* posin = s.f((size_t)CH * kn * 3);
    float* pos   = s.f((size_t)CH * kn * D);
    float* b1    = s.f((size_t)CH * kn * D);
    float* b2    = s.f((size_t)CH * kn * D);

    for (int r0 = 0; r0 < n; r0 += CH) {
        int rows = (n - r0 < CH) ? (n - r0) : CH;
        int Mr = rows * kn;
        const int* idxc = idx + (size_t)r0 * kn;
        EW(build_posin, (long)Mr * 3, xyz, idxc, r0, rows, kn, posin);
        gemmL(st, posin, t.d1, b1, Mr, D, 3, 1);            // relu(delta1)
        gemmL(st, b1, t.d2, pos, Mr, D, D, 0);              // delta2 -> pos
        EW(tb_prep, (long)Mr * D, qb, kb, pos, idxc, r0, rows, kn, D, b1);
        gemmL(st, b1, t.g1, b2, Mr, D, D, 1);               // relu(gamma1)
        gemmL(st, b2, t.g2, b1, Mr, D, D, 0);               // gamma2 -> attn logits
        EW(attn_agg, (long)rows * D, b1, pos, vb, idxc, r0, rows, kn, D, scale, agg);
    }
    gemmL(st, agg, t.fc2, pts, n, C, D, 0, nullptr, nullptr, pts);  // + residual
}

extern "C" void kernel_launch(void* const* d_in, const int* in_sizes, int n_in,
                              void* d_out, int out_size, void* d_ws, size_t ws_size,
                              hipStream_t stream)
{
    (void)in_sizes; (void)n_in; (void)out_size;
    hipStream_t st = stream;

    // ---- unpack params in JAX pytree order (dict keys sorted) ----
    int p = 0;
    const float* x = FP(d_in, p);                 // (4000, 3)
    Lin fc1a = rdLin(d_in, p), fc1b = rdLin(d_in, p);
    Lin fc2a = rdLin(d_in, p), fc2b = rdLin(d_in, p), fc2c = rdLin(d_in, p);
    Lin fc3a = rdLin(d_in, p), fc3b = rdLin(d_in, p), fc3c = rdLin(d_in, p);
    TB t1 = rdTB(d_in, p);
    TB t2 = rdTB(d_in, p);
    SAL td[4][2];
    for (int i = 0; i < 4; ++i) { td[i][0] = rdSAL(d_in, p); td[i][1] = rdSAL(d_in, p); }
    TB tfd[4]; for (int i = 0; i < 4; ++i) tfd[i] = rdTB(d_in, p);
    TB tfu[4]; for (int i = 0; i < 4; ++i) tfu[i] = rdTB(d_in, p);
    TU tu[4];  for (int i = 0; i < 4; ++i) tu[i]  = rdTU(d_in, p);

    // ---- workspace layout ----
    Bump ws{(char*)d_ws, 0, ws_size};
    const int npl[5] = {4000, 1000, 250, 62, 15};
    const int cl[5]  = {32, 64, 128, 256, 512};

    const float* lxyz[5];
    float* lpts[5];
    lxyz[0] = x;
    lpts[0] = ws.f((size_t)npl[0] * cl[0]);
    for (int l = 1; l < 5; ++l) {
        lxyz[l] = ws.f((size_t)npl[l] * 3);
        lpts[l] = ws.f((size_t)npl[l] * cl[l]);
    }
    float* ping = ws.f((size_t)4000 * 512);
    float* pong = ws.f((size_t)4000 * 512);

    // ---- stem: pts = fc1b(relu(fc1a(x))) ; transformer t1 ----
    {
        Bump s = ws;
        float* tmp = s.f((size_t)4000 * 32);
        gemmL(st, x, fc1a, tmp, 4000, 32, 3, 1);
        gemmL(st, tmp, fc1b, lpts[0], 4000, 32, 32, 0);
        run_tblock(st, s, t1, lxyz[0], lpts[0], 4000, 32, 16);
    }

    // ---- encoder: set abstraction + transformer per level ----
    for (int i = 0; i < 4; ++i) {
        int nN = npl[i], nS = npl[i + 1];
        int Cin = cl[i], Cc = cl[i + 1];
        int cin = Cin + 3;
        Bump s = ws;

        int* fidx = s.i(nS);
        fps_kernel<<<1, 1024, 0, st>>>(lxyz[i], nN, nS, fidx);
        gather_rows<<<cdiv((long)nS * 3, 256), 256, 0, st>>>(lxyz[i], fidx, nS, 3,
                                                             (float*)lxyz[i + 1]);
        int* idx = s.i((size_t)nS * 16);
        knn_kernel<<<cdiv(nS, 128), 128, 0, st>>>(lxyz[i + 1], nS, lxyz[i], nN, 16, idx, nullptr);

        float* h  = s.f((size_t)nS * 16 * cin);
        float* h2 = s.f((size_t)nS * 16 * Cc);
        float* h3 = s.f((size_t)nS * 16 * Cc);
        EW(build_sa_h, (long)nS * 16 * cin, lxyz[i], lxyz[i + 1], lpts[i], idx, nS, 16, Cin, h);
        gemmL(st, h,  td[i][0].lin, h2, nS * 16, Cc, cin, 1, td[i][0].gamma, td[i][0].beta);
        gemmL(st, h2, td[i][1].lin, h3, nS * 16, Cc, Cc,  1, td[i][1].gamma, td[i][1].beta);
        EW(maxpool_k, (long)nS * Cc, h3, nS, 16, Cc, lpts[i + 1]);

        int kn = (nS < 16) ? nS : 16;
        run_tblock(st, s, tfd[i], lxyz[i + 1], lpts[i + 1], nS, Cc, kn);
    }

    // ---- bottleneck MLP + transformer t2 (n=15) ----
    {
        Bump s = ws;
        float* m1 = s.f((size_t)15 * 512);
        float* m2 = s.f((size_t)15 * 512);
        gemmL(st, lpts[4], fc2a, m1, 15, 512, 512, 1);
        gemmL(st, m1, fc2b, m2, 15, 512, 512, 1);
        gemmL(st, m2, fc2c, lpts[4], 15, 512, 512, 0);
        run_tblock(st, s, t2, lxyz[4], lpts[4], 15, 512, 15);
    }

    // ---- decoder: transition-up + transformer per level ----
    float* cur = lpts[4];
    for (int j = 0; j < 4; ++j) {
        int lc = 4 - j, lf = 3 - j;
        int nC = npl[lc], nF = npl[lf];
        int c = cl[lf];
        Bump s = ws;

        float* f1 = s.f((size_t)nC * c);
        float* f2 = s.f((size_t)nF * c);
        gemmL(st, cur, tu[j].fc1, f1, nC, c, 2 * c, 1, tu[j].bn1g, tu[j].bn1b);
        gemmL(st, lpts[lf], tu[j].fc2, f2, nF, c, c, 1, tu[j].bn2g, tu[j].bn2b);

        int* i3 = s.i((size_t)nF * 3);
        float* d3 = s.f((size_t)nF * 3);
        knn_kernel<<<cdiv(nF, 128), 128, 0, st>>>(lxyz[lf], nF, lxyz[lc], nC, 3, i3, d3);

        float* dst = (j % 2 == 0) ? ping : pong;
        EW(interp3_k, (long)nF * c, i3, d3, f1, f2, nF, c, dst);

        int kn = (nF < 16) ? nF : 16;
        run_tblock(st, s, tfu[j], lxyz[lf], dst, nF, c, kn);
        cur = dst;
    }

    // ---- head: fc3c(relu(fc3b(relu(fc3a(pts))))) -> d_out (4000, 6) ----
    {
        Bump s = ws;
        float* hA = s.f((size_t)4000 * 64);
        float* hB = s.f((size_t)4000 * 64);
        gemmL(st, cur, fc3a, hA, 4000, 64, 32, 1);
        gemmL(st, hA, fc3b, hB, 4000, 64, 64, 1);
        gemmL(st, hB, fc3c, (float*)d_out, 4000, 6, 64, 0);
    }
}
